// TransformerEncoder_41506563948546
// MI455X (gfx1250) — compile-verified, matrix operand order checked by reading, not run on hardware
//
#include <hip/hip_runtime.h>

// ---------------------------------------------------------------------------
// Transformer encoder (B=8,S=512,H=768,NH=12,D=64,L=12) for gfx1250 (CDNA5).
// bf16 WMMA everywhere; async-LDS DMA for A tiles; weights pre-transposed bf16.
// ---------------------------------------------------------------------------

typedef __bf16 bf16;
typedef __attribute__((ext_vector_type(16))) __bf16 v16bf;
typedef __attribute__((ext_vector_type(8)))  __bf16 v8bf;
typedef __attribute__((ext_vector_type(8)))  float  v8f;

#define B_   8
#define S_   512
#define H_   768
#define NH_  12
#define D_   64
#define L_   12
#define MTOT (B_ * S_)      // 4096
#define KSTEPS (H_ / 32)    // 24

// GEMM tiling
#define BM  128
#define BN  64
#define LDA 40              // padded LDS row (bf16 elems): 80B, 16B aligned, conflict-free
#define LDV 40
#define LDP 40

static __device__ inline v8f wmma_bf16(v16bf a, v16bf b, v8f c) {
  return __builtin_amdgcn_wmma_f32_16x16x32_bf16(false, a, false, b, (short)0, c,
                                                 false, false);
}

static __device__ inline v8f zero_v8f() {
  v8f z;
#pragma unroll
  for (int i = 0; i < 8; ++i) z[i] = 0.0f;
  return z;
}

// 16x32 bf16 fragment (A-style layout): lane = m+16g holds row m,
// K elems [8g,8g+8) in f[0..7], [16+8g,16+8g+8) in f[8..15].
// Works for LDS (ds_load_b128) and global (global_load_b128) pointers.
static __device__ inline v16bf ld_frag(const bf16* rowbase, int g) {
  const v8bf lo = *reinterpret_cast<const v8bf*>(rowbase + 8 * g);
  const v8bf hi = *reinterpret_cast<const v8bf*>(rowbase + 16 + 8 * g);
  v16bf f;
#pragma unroll
  for (int i = 0; i < 8; ++i) { f[i] = lo[i]; f[i + 8] = hi[i]; }
  return f;
}

// Async DMA: 16B global -> LDS per lane (ASYNCcnt-tracked, no VGPR round trip)
static __device__ inline void async_copy16(unsigned lds_off, const bf16* gptr) {
  asm volatile("global_load_async_to_lds_b128 %0, %1, off"
               :: "v"(lds_off), "v"((unsigned long long)(uintptr_t)gptr)
               : "memory");
}

// ---------------------------------------------------------------------------
// GEMM: C = act(A[4096,768](bf16) @ Wt^T + bias),  Wt is [N][K] bf16 (pre-T)
// Cf: optional fp32 out; Cb: optional bf16 out (scaled by oscale).
// ---------------------------------------------------------------------------
__global__ __launch_bounds__(256)
void gemm_bf16_kernel(const bf16* __restrict__ A, const bf16* __restrict__ Wt,
                      const float* __restrict__ bias, float* __restrict__ Cf,
                      bf16* __restrict__ Cb, int relu, float oscale) {
  __shared__ __align__(16) bf16 As[2][BM * LDA];

  const int t    = threadIdx.x;
  const int lane = t & 31;
  const int w    = t >> 5;            // 8 waves: 4x2
  const int wm   = w >> 1;
  const int wn   = w & 1;
  const int bm   = blockIdx.y * BM;
  const int bn   = blockIdx.x * BN;
  const int m_   = lane & 15;
  const int g    = lane >> 4;

  // per-thread async staging addresses: 2 chunks of 16B; row = idx>>2, ch = idx&3
  const int r0 = t >> 2,           c0 = t & 3;
  const int r1 = (256 + t) >> 2,   c1 = (256 + t) & 3;

  auto stage = [&](int buf, int k0) {
    async_copy16((unsigned)(uintptr_t)&As[buf][r0 * LDA + c0 * 8],
                 A + (size_t)(bm + r0) * H_ + k0 + c0 * 8);
    async_copy16((unsigned)(uintptr_t)&As[buf][r1 * LDA + c1 * 8],
                 A + (size_t)(bm + r1) * H_ + k0 + c1 * 8);
  };

  // named accumulators so accXY = wmma(a,b,accXY) coalesces in place
  v8f acc00 = zero_v8f(), acc01 = zero_v8f();
  v8f acc10 = zero_v8f(), acc11 = zero_v8f();

  const bf16* wrow0 = Wt + (size_t)(bn + wn * 32 + m_) * H_;
  const bf16* wrow1 = Wt + (size_t)(bn + wn * 32 + 16 + m_) * H_;
  const bf16* arow0;
  const bf16* arow1;

  auto compute = [&](int buf, int k0) {
    const bf16* as = As[buf];
    v16bf a0 = ld_frag(as + (wm * 32 + m_) * LDA, g);
    v16bf a1 = ld_frag(as + (wm * 32 + 16 + m_) * LDA, g);
    v16bf b0 = ld_frag(wrow0 + k0, g);   // straight from global (L2-resident)
    v16bf b1 = ld_frag(wrow1 + k0, g);
    acc00 = wmma_bf16(a0, b0, acc00);
    acc01 = wmma_bf16(a0, b1, acc01);
    acc10 = wmma_bf16(a1, b0, acc10);
    acc11 = wmma_bf16(a1, b1, acc11);
  };

  stage(0, 0);
  for (int kk = 0; kk < KSTEPS - 1; ++kk) {
    const int cur = kk & 1;
    stage(cur ^ 1, (kk + 1) * 32);
    asm volatile("s_wait_asynccnt 0x2" ::: "memory");   // old buffer landed
    __syncthreads();
    compute(cur, kk * 32);
    __syncthreads();
  }
  asm volatile("s_wait_asynccnt 0x0" ::: "memory");
  __syncthreads();
  compute((KSTEPS - 1) & 1, (KSTEPS - 1) * 32);

  const v8f* accs[2][2] = {{&acc00, &acc01}, {&acc10, &acc11}};
#pragma unroll
  for (int tm = 0; tm < 2; ++tm) {
#pragma unroll
    for (int tn = 0; tn < 2; ++tn) {
      const int col = bn + wn * 32 + tn * 16 + m_;
      const float bv = bias[col];
      const v8f av = *accs[tm][tn];
#pragma unroll
      for (int r = 0; r < 8; ++r) {
        const int row = bm + wm * 32 + tm * 16 + r + 8 * g;
        float val = av[r] + bv;
        if (relu) val = fmaxf(val, 0.0f);
        const size_t idx = (size_t)row * H_ + col;
        if (Cf) Cf[idx] = val;
        if (Cb) Cb[idx] = (bf16)(val * oscale);
      }
    }
  }
}

// ---------------------------------------------------------------------------
// Flash attention: ctx = softmax(q k^T) v (1/sqrt(D) pre-folded into q).
// q,k,v,ctx all bf16. Block = (b,h,128 q rows); wave = 16 q rows.
// ---------------------------------------------------------------------------
__global__ __launch_bounds__(256)
void attention_kernel(const bf16* __restrict__ q, const bf16* __restrict__ k,
                      const bf16* __restrict__ v, bf16* __restrict__ ctx) {
  __shared__ __align__(16) bf16 Vt[64 * LDV];       // [d][key] transposed
  __shared__ __align__(16) bf16 Pw[8][16 * LDP];    // per-wave probs

  const int t    = threadIdx.x;
  const int lane = t & 31;
  const int w    = t >> 5;
  const int blk  = blockIdx.x;
  const int qb   = blk & 3;
  const int h    = (blk >> 2) % NH_;
  const int b    = blk / (4 * NH_);
  const int m_   = lane & 15;
  const int g    = lane >> 4;
  const int qrow0 = qb * 128 + w * 16;

  const bf16* qrow = q + ((size_t)(b * S_) + qrow0 + m_) * H_ + h * D_;
  v16bf qa0 = ld_frag(qrow, g);          // d 0..31
  v16bf qa1 = ld_frag(qrow + 32, g);     // d 32..63

  v8f acc0 = zero_v8f(), acc1 = zero_v8f(), acc2 = zero_v8f(), acc3 = zero_v8f();
  float mrow[8], lrow[8];
#pragma unroll
  for (int r = 0; r < 8; ++r) { mrow[r] = -3.0e38f; lrow[r] = 0.0f; }

  // per-thread V staging coords: 8 bf16 (16B) per thread per chunk
  const int vkey = t >> 3, vd = (t & 7) * 8;

  for (int c = 0; c < S_ / 32; ++c) {
    const int ks = c * 32;
    // stage V chunk transposed [d][key]
    {
      const v8bf vv = *reinterpret_cast<const v8bf*>(
          v + ((size_t)(b * S_) + ks + vkey) * H_ + h * D_ + vd);
#pragma unroll
      for (int j = 0; j < 8; ++j) Vt[(vd + j) * LDV + vkey] = vv[j];
    }
    __syncthreads();

    // scores: two 16x16 tiles; K fragments straight from global bf16
    const bf16* krow0 = k + ((size_t)(b * S_) + ks + m_) * H_ + h * D_;
    const bf16* krow1 = krow0 + (size_t)16 * H_;
    v8f s0 = zero_v8f(), s1 = zero_v8f();
    s0 = wmma_bf16(qa0, ld_frag(krow0, g), s0);
    s0 = wmma_bf16(qa1, ld_frag(krow0 + 32, g), s0);
    s1 = wmma_bf16(qa0, ld_frag(krow1, g), s1);
    s1 = wmma_bf16(qa1, ld_frag(krow1 + 32, g), s1);

    // online softmax (row r+8g lives in 16-lane half g)
#pragma unroll
    for (int r = 0; r < 8; ++r) {
      float x = fmaxf(s0[r], s1[r]);
      x = fmaxf(x, __shfl_xor(x, 1, 32));
      x = fmaxf(x, __shfl_xor(x, 2, 32));
      x = fmaxf(x, __shfl_xor(x, 4, 32));
      x = fmaxf(x, __shfl_xor(x, 8, 32));
      const float mnew  = fmaxf(mrow[r], x);
      const float scale = __expf(mrow[r] - mnew);
      const float p0 = __expf(s0[r] - mnew);
      const float p1 = __expf(s1[r] - mnew);
      float rs = p0 + p1;
      rs += __shfl_xor(rs, 1, 32);
      rs += __shfl_xor(rs, 2, 32);
      rs += __shfl_xor(rs, 4, 32);
      rs += __shfl_xor(rs, 8, 32);
      lrow[r] = lrow[r] * scale + rs;
      mrow[r] = mnew;
      acc0[r] *= scale; acc1[r] *= scale; acc2[r] *= scale; acc3[r] *= scale;
      bf16* pr = Pw[w] + (r + 8 * g) * LDP;   // same-wave DS ops are in-order
      pr[m_]      = (bf16)p0;
      pr[16 + m_] = (bf16)p1;
    }

    v16bf pa = ld_frag(Pw[w] + m_ * LDP, g);
    acc0 = wmma_bf16(pa, ld_frag(Vt + (0 * 16 + m_) * LDV, g), acc0);
    acc1 = wmma_bf16(pa, ld_frag(Vt + (1 * 16 + m_) * LDV, g), acc1);
    acc2 = wmma_bf16(pa, ld_frag(Vt + (2 * 16 + m_) * LDV, g), acc2);
    acc3 = wmma_bf16(pa, ld_frag(Vt + (3 * 16 + m_) * LDV, g), acc3);
    __syncthreads();
  }

#pragma unroll
  for (int r = 0; r < 8; ++r) {
    const float inv = 1.0f / lrow[r];
    bf16* dst = ctx + ((size_t)(b * S_) + qrow0 + r + 8 * g) * H_ + h * D_;
    dst[0 * 16 + m_] = (bf16)(acc0[r] * inv);
    dst[1 * 16 + m_] = (bf16)(acc1[r] * inv);
    dst[2 * 16 + m_] = (bf16)(acc2[r] * inv);
    dst[3 * 16 + m_] = (bf16)(acc3[r] * inv);
  }
}

// ---------------------------------------------------------------------------
// out = LayerNorm(x + res) * gamma + beta; optional bf16 copy for next GEMM.
// ---------------------------------------------------------------------------
__global__ __launch_bounds__(256)
void add_ln_kernel(const float* __restrict__ x, const float* __restrict__ res,
                   const float* __restrict__ gamma, const float* __restrict__ beta,
                   float* __restrict__ out, bf16* __restrict__ outb) {
  __shared__ float red[256];
  const int t = threadIdx.x;
  const size_t row = blockIdx.x;
  const float* px = x + row * H_;
  const float* pr = res + row * H_;

  float vals[3];
  float s = 0.0f;
#pragma unroll
  for (int i = 0; i < 3; ++i) {
    vals[i] = px[t + i * 256] + pr[t + i * 256];
    s += vals[i];
  }
  red[t] = s;
  __syncthreads();
  for (int off = 128; off > 0; off >>= 1) {
    if (t < off) red[t] += red[t + off];
    __syncthreads();
  }
  const float mean = red[0] * (1.0f / H_);
  __syncthreads();

  float vs = 0.0f;
#pragma unroll
  for (int i = 0; i < 3; ++i) {
    const float d = vals[i] - mean;
    vs += d * d;
  }
  red[t] = vs;
  __syncthreads();
  for (int off = 128; off > 0; off >>= 1) {
    if (t < off) red[t] += red[t + off];
    __syncthreads();
  }
  const float inv = rsqrtf(red[0] * (1.0f / H_) + 1e-12f);

#pragma unroll
  for (int i = 0; i < 3; ++i) {
    const int c = t + i * 256;
    const float o = (vals[i] - mean) * inv * gamma[c] + beta[c];
    out[row * H_ + c] = o;
    if (outb) outb[row * H_ + c] = (bf16)o;
  }
}

// ---------------------------------------------------------------------------
// Per-layer weight prep: fp32 [K][N] -> bf16 transposed [N][K], 5 matrices.
// ---------------------------------------------------------------------------
__global__ __launch_bounds__(256)
void convert_wt_kernel(const float* __restrict__ Wq, const float* __restrict__ Wk,
                       const float* __restrict__ Wv, const float* __restrict__ Wo,
                       const float* __restrict__ W2, bf16* __restrict__ dst) {
  __shared__ float tile[32][33];
  const int z = blockIdx.z;
  const float* src = (z == 0) ? Wq : (z == 1) ? Wk : (z == 2) ? Wv
                   : (z == 3) ? Wo : W2;
  bf16* d = dst + (size_t)z * H_ * H_;
  const int tx = threadIdx.x & 31;
  const int ty = threadIdx.x >> 5;   // 0..7
  const int n0 = blockIdx.x * 32, k0 = blockIdx.y * 32;
#pragma unroll
  for (int j = 0; j < 4; ++j)
    tile[ty + 8 * j][tx] = src[(size_t)(k0 + ty + 8 * j) * H_ + n0 + tx];
  __syncthreads();
#pragma unroll
  for (int j = 0; j < 4; ++j)
    d[(size_t)(n0 + ty + 8 * j) * H_ + k0 + tx] = (bf16)tile[tx][ty + 8 * j];
}

__global__ __launch_bounds__(256)
void convert_x_kernel(const float* __restrict__ src, bf16* __restrict__ dst) {
  const size_t i = (size_t)blockIdx.x * 1024 + threadIdx.x * 4;
  const float4 vv = *reinterpret_cast<const float4*>(src + i);
  dst[i + 0] = (bf16)vv.x; dst[i + 1] = (bf16)vv.y;
  dst[i + 2] = (bf16)vv.z; dst[i + 3] = (bf16)vv.w;
}

// ---------------------------------------------------------------------------
// Driver
// ---------------------------------------------------------------------------
extern "C" void kernel_launch(void* const* d_in, const int* in_sizes, int n_in,
                              void* d_out, int out_size, void* d_ws, size_t ws_size,
                              hipStream_t stream) {
  const float* x    = (const float*)d_in[0];
  const float* Wq   = (const float*)d_in[1];
  const float* bq   = (const float*)d_in[2];
  const float* Wk   = (const float*)d_in[3];
  const float* bk   = (const float*)d_in[4];
  const float* Wv   = (const float*)d_in[5];
  const float* bv   = (const float*)d_in[6];
  const float* Wo   = (const float*)d_in[7];
  const float* bo   = (const float*)d_in[8];
  const float* g1   = (const float*)d_in[9];
  const float* be1  = (const float*)d_in[10];
  const float* W2   = (const float*)d_in[11];
  const float* b2   = (const float*)d_in[12];
  const float* g2   = (const float*)d_in[13];
  const float* be2  = (const float*)d_in[14];

  float* out = (float*)d_out;                  // fp32 x carried across layers
  const size_t n  = (size_t)MTOT * H_;
  const size_t hh = (size_t)H_ * H_;

  bf16* xb    = (bf16*)d_ws;
  bf16* qb16  = xb + n;
  bf16* kb16  = qb16 + n;
  bf16* vb16  = kb16 + n;
  bf16* ctxb  = vb16 + n;
  bf16* h1b   = ctxb + n;
  bf16* out1b = h1b + n;
  bf16* wt    = out1b + n;                     // 5*H*H bf16 (per-layer, transposed)
  float* out1 = (float*)(wt + 5 * hh);
  float* tmp  = out1 + n;

  bf16* wqt = wt + 0 * hh;
  bf16* wkt = wt + 1 * hh;
  bf16* wvt = wt + 2 * hh;
  bf16* wot = wt + 3 * hh;
  bf16* w2t = wt + 4 * hh;

  hipMemcpyAsync(out, x, n * sizeof(float), hipMemcpyDeviceToDevice, stream);

  const dim3 blk256(256);
  const dim3 gemm_grid(H_ / BN, MTOT / BM);    // (12, 32)
  const dim3 attn_grid(B_ * NH_ * (S_ / 128)); // 384
  const dim3 ln_grid(MTOT);
  const dim3 wt_grid(H_ / 32, H_ / 32, 5);     // (24,24,5)

  convert_x_kernel<<<dim3((unsigned)(n / 1024)), blk256, 0, stream>>>(x, xb);

  for (int l = 0; l < L_; ++l) {
    const size_t wofs = (size_t)l * hh;
    const size_t vofs = (size_t)l * H_;

    convert_wt_kernel<<<wt_grid, blk256, 0, stream>>>(
        Wq + wofs, Wk + wofs, Wv + wofs, Wo + wofs, W2 + wofs, wt);

    gemm_bf16_kernel<<<gemm_grid, blk256, 0, stream>>>(
        xb, wqt, bq + vofs, nullptr, qb16, 0, 0.125f);   // fold 1/sqrt(D)
    gemm_bf16_kernel<<<gemm_grid, blk256, 0, stream>>>(
        xb, wkt, bk + vofs, nullptr, kb16, 0, 1.0f);
    gemm_bf16_kernel<<<gemm_grid, blk256, 0, stream>>>(
        xb, wvt, bv + vofs, nullptr, vb16, 0, 1.0f);

    attention_kernel<<<attn_grid, blk256, 0, stream>>>(qb16, kb16, vb16, ctxb);

    gemm_bf16_kernel<<<gemm_grid, blk256, 0, stream>>>(
        ctxb, wot, bo + vofs, tmp, nullptr, 0, 1.0f);
    add_ln_kernel<<<ln_grid, blk256, 0, stream>>>(
        out, tmp, g1 + vofs, be1 + vofs, out1, out1b);

    gemm_bf16_kernel<<<gemm_grid, blk256, 0, stream>>>(
        out1b, w2t, b2 + vofs, nullptr, h1b, 1, 1.0f);
    gemm_bf16_kernel<<<gemm_grid, blk256, 0, stream>>>(
        h1b, w2t, b2 + vofs, tmp, nullptr, 1, 1.0f);
    add_ln_kernel<<<ln_grid, blk256, 0, stream>>>(
        out1, tmp, g2 + vofs, be2 + vofs, out, xb);
  }
}
